// HybridParallelSequentialExtractor_29970281791929
// MI455X (gfx1250) — compile-verified, matrix operand order checked by reading, not run on hardware
//
#include <hip/hip_runtime.h>
#include <hip/hip_bf16.h>
#include <math.h>

// ---------------------------------------------------------------------------
// HybridParallelSequentialExtractor forward for MI455X (gfx1250, wave32)
// GEMM-shaped stages (kNN distance matrices, EdgeConv 1x1 convs, fusion GEMM)
// use v_wmma_f32_16x16x32_f16 with vectorized (b128) LDS fragment loads;
// irregular stages (KPConv, PointTransformer, LN, softmax, pooling) are VALU.
// ---------------------------------------------------------------------------

typedef __attribute__((ext_vector_type(16))) _Float16 v16h;
typedef __attribute__((ext_vector_type(8)))  _Float16 v8h;
typedef __attribute__((ext_vector_type(8)))  float    v8f;

#define BATCH 2
#define NPTS  4096
#define KSEL  20      // DGCNN knn
#define KNBR  16      // KPConv / PT knn (prefix of KSEL list on pos)
#define NKP   15
#define LN_EPS 1e-5f
#define SLOPE  0.2f

// ---- WMMA fragment loader (16x16x32 f16), ISA 7.12.2 layout --------------
// Lane -> M(row)/N(col) = row_base + (lane&15); element e -> K:
//   k = kbase + ((lane>=16)?8:0) + (e/8)*16 + (e%8)
// Each half of the fragment is 8 contiguous halfwords => two b128 LDS loads.
__device__ inline v16h frag_ld(const _Float16* buf, int row_base, int ld,
                               int kbase, int lane) {
  const int m    = row_base + (lane & 15);
  const int koff = kbase + ((lane >> 4) << 3);
  const v8h lo = *(const v8h*)(buf + m * ld + koff);
  const v8h hi = *(const v8h*)(buf + m * ld + koff + 16);
  return __builtin_shufflevector(lo, hi, 0, 1, 2, 3, 4, 5, 6, 7,
                                 8, 9, 10, 11, 12, 13, 14, 15);
}

#define WMMA(a, b, c) __builtin_amdgcn_wmma_f32_16x16x32_f16( \
    false, (a), false, (b), (short)0, (c), false, false)

// ---------------------------------------------------------------------------
// Kernel 1: normalize point cloud.  x [B,3,N] -> pos [B,N,3]
// ---------------------------------------------------------------------------
__global__ __launch_bounds__(256)
void normalize_kernel(const float* __restrict__ x, float* __restrict__ pos) {
  __shared__ float red[256];
  __shared__ float mean[3];
  __shared__ float maxd;
  const int b = blockIdx.x, tid = threadIdx.x;
  for (int c = 0; c < 3; ++c) {
    float s = 0.f;
    for (int n = tid; n < NPTS; n += 256) s += x[((size_t)b * 3 + c) * NPTS + n];
    red[tid] = s; __syncthreads();
    for (int o = 128; o > 0; o >>= 1) {
      if (tid < o) red[tid] += red[tid + o];
      __syncthreads();
    }
    if (tid == 0) mean[c] = red[0] / (float)NPTS;
    __syncthreads();
  }
  float mx = 0.f;
  for (int n = tid; n < NPTS; n += 256) {
    float dx = x[((size_t)b * 3 + 0) * NPTS + n] - mean[0];
    float dy = x[((size_t)b * 3 + 1) * NPTS + n] - mean[1];
    float dz = x[((size_t)b * 3 + 2) * NPTS + n] - mean[2];
    mx = fmaxf(mx, sqrtf(dx * dx + dy * dy + dz * dz));
  }
  red[tid] = mx; __syncthreads();
  for (int o = 128; o > 0; o >>= 1) {
    if (tid < o) red[tid] = fmaxf(red[tid], red[tid + o]);
    __syncthreads();
  }
  if (tid == 0) maxd = red[0];
  __syncthreads();
  const float inv = 1.f / (maxd + 1e-8f);
  for (int n = tid; n < NPTS; n += 256)
    for (int c = 0; c < 3; ++c)
      pos[((size_t)b * NPTS + n) * 3 + c] =
          (x[((size_t)b * 3 + c) * NPTS + n] - mean[c]) * inv;
}

// ---------------------------------------------------------------------------
// Kernel 2: WMMA kNN (templated on feature dim C). One wave = 32 query rows;
// loops over 16-point column tiles, 16x16 (x2) inner product via WMMA,
// converts to -||q-p||^2, streaming per-row top-KSEL selection.
// ---------------------------------------------------------------------------
template <int C>
__global__ __launch_bounds__(32)
void knn_kernel(const float* __restrict__ feat, int fstride,
                int* __restrict__ idx_out, int idx_stride) {
  constexpr int KP  = (C + 31) & ~31;      // padded K (<=64)
  constexpr int NKC = KP >> 5;
  __shared__ __align__(16) _Float16 Qh[32 * KP];
  __shared__ __align__(16) _Float16 Ph[16 * KP];
  __shared__ float qn[32], pn[16];
  __shared__ float dt[32 * 16];
  const int lane = threadIdx.x;
  const int tiles = NPTS / 32;
  const int b = blockIdx.x / tiles;
  const int q0 = (blockIdx.x % tiles) * 32;
  // stage 32 query rows (f16, zero padded) + squared norms
  {
    const float* qp = feat + ((size_t)b * NPTS + q0 + lane) * fstride;
    float s = 0.f;
#pragma unroll
    for (int c = 0; c < KP; ++c) {
      float v = (c < C) ? qp[c] : 0.f;
      Qh[lane * KP + c] = (_Float16)v;
      s += v * v;
    }
    qn[lane] = s;
  }
  __syncthreads();
  v16h afr[2][NKC];
#pragma unroll
  for (int m = 0; m < 2; ++m)
#pragma unroll
    for (int kc = 0; kc < NKC; ++kc)
      afr[m][kc] = frag_ld(Qh, m * 16, KP, kc * 32, lane);

  float bd[KSEL];
  int   bi[KSEL];
#pragma unroll
  for (int i = 0; i < KSEL; ++i) { bd[i] = -3.4e38f; bi[i] = 0; }

  for (int t = 0; t < NPTS / 16; ++t) {
    __syncthreads();                       // WAR on Ph / dt
    if (lane < 16) {
      const float* pp = feat + ((size_t)b * NPTS + t * 16 + lane) * fstride;
      float s = 0.f;
#pragma unroll
      for (int c = 0; c < KP; ++c) {
        float v = (c < C) ? pp[c] : 0.f;
        Ph[lane * KP + c] = (_Float16)v;
        s += v * v;
      }
      pn[lane] = s;
    }
    __syncthreads();
    v8f acc0 = {}, acc1 = {};
#pragma unroll
    for (int kc = 0; kc < NKC; ++kc) {
      v16h bf = frag_ld(Ph, 0, KP, kc * 32, lane);
      acc0 = WMMA(afr[0][kc], bf, acc0);
      acc1 = WMMA(afr[1][kc], bf, acc1);
    }
    const int col = lane & 15;
    const int rb  = (lane >> 4) << 3;
#pragma unroll
    for (int r = 0; r < 8; ++r) {
      int m0 = rb + r;
      dt[m0 * 16 + col]        = 2.f * acc0[r] - qn[m0]      - pn[col];
      dt[(16 + m0) * 16 + col] = 2.f * acc1[r] - qn[16 + m0] - pn[col];
    }
    __syncthreads();
    // each lane owns one query row: insertion into sorted top-KSEL
    for (int c = 0; c < 16; ++c) {
      float d = dt[lane * 16 + c];
      int j = t * 16 + c;
      if (d > bd[KSEL - 1]) {
        int p = KSEL - 1;
        while (p > 0 && bd[p - 1] < d) {
          bd[p] = bd[p - 1]; bi[p] = bi[p - 1]; --p;
        }
        bd[p] = d; bi[p] = j;
      }
    }
  }
  int* op = idx_out + ((size_t)b * NPTS + q0 + lane) * idx_stride;
#pragma unroll
  for (int kk = 0; kk < KSEL; ++kk) op[kk] = bi[kk];
}

// ---------------------------------------------------------------------------
// Kernel 3: EdgeConv: graph_feature + 1x1 conv (WMMA) + LayerNorm +
// LeakyReLU + max over k + residual.  One wave per point.
// RMODE: 0 none, 1 identity residual, 2 residual @ align^T (Cres == Cin)
// ---------------------------------------------------------------------------
template <int Cin, int Oc, int RMODE>
__global__ __launch_bounds__(32)
void edgeconv_kernel(const float* __restrict__ in, int istride,
                     const int* __restrict__ idx, int idx_stride,
                     const float* __restrict__ W, const float* __restrict__ gamma,
                     const float* __restrict__ beta,
                     const float* __restrict__ res, int rstride,
                     const float* __restrict__ align,
                     float* __restrict__ out, int ostride) {
  constexpr int K2 = 2 * Cin;
  constexpr int KP = (K2 + 31) & ~31;
  constexpr int NKC = KP >> 5;
  constexpr int NT  = Oc >> 4;
  __shared__ __align__(16) _Float16 E[32 * KP];   // edge feats [k(pad32)][2C]
  __shared__ __align__(16) _Float16 Wh[Oc * KP];  // weights as [o][c]
  __shared__ float dmat[32 * Oc];                 // conv output [k][O]
  __shared__ float ctr[Cin];
  const int lane = threadIdx.x;
  const int n = blockIdx.x % NPTS, b = blockIdx.x / NPTS;
  const size_t pbase = (size_t)b * NPTS + n;

  for (int c = lane; c < Cin; c += 32) ctr[c] = in[pbase * istride + c];
  for (int e = lane; e < Oc * KP; e += 32) {
    int o = e / KP, c = e % KP;
    Wh[o * KP + c] = (_Float16)((c < K2) ? W[o * K2 + c] : 0.f);
  }
  __syncthreads();
  {
    int r = lane;
    if (r < KSEL) {
      int nb = idx[pbase * idx_stride + r];
      const float* np = in + ((size_t)b * NPTS + nb) * istride;
#pragma unroll
      for (int c = 0; c < Cin; ++c) {
        float cv = ctr[c];
        E[r * KP + c]       = (_Float16)(np[c] - cv);
        E[r * KP + Cin + c] = (_Float16)cv;
      }
#pragma unroll
      for (int c = K2; c < KP; ++c) E[r * KP + c] = (_Float16)0.f;
    } else {
#pragma unroll
      for (int c = 0; c < KP; ++c) E[r * KP + c] = (_Float16)0.f;
    }
  }
  __syncthreads();
#pragma unroll
  for (int ot = 0; ot < NT; ++ot) {
    v8f acc0 = {}, acc1 = {};
#pragma unroll
    for (int kc = 0; kc < NKC; ++kc) {
      v16h a0 = frag_ld(E, 0, KP, kc * 32, lane);
      v16h a1 = frag_ld(E, 16, KP, kc * 32, lane);
      v16h bf = frag_ld(Wh, ot * 16, KP, kc * 32, lane);
      acc0 = WMMA(a0, bf, acc0);
      acc1 = WMMA(a1, bf, acc1);
    }
    const int col = ot * 16 + (lane & 15);
    const int rb  = (lane >> 4) << 3;
#pragma unroll
    for (int r = 0; r < 8; ++r) {
      dmat[(rb + r) * Oc + col]      = acc0[r];
      dmat[(16 + rb + r) * Oc + col] = acc1[r];
    }
  }
  __syncthreads();
  if (lane < KSEL) {       // LayerNorm over channels + LeakyReLU, per row
    float mu = 0.f, m2 = 0.f;
#pragma unroll
    for (int o = 0; o < Oc; ++o) { float v = dmat[lane * Oc + o]; mu += v; m2 += v * v; }
    mu /= (float)Oc; m2 = m2 / (float)Oc - mu * mu;
    float inv = rsqrtf(m2 + LN_EPS);
#pragma unroll
    for (int o = 0; o < Oc; ++o) {
      float v = (dmat[lane * Oc + o] - mu) * inv * gamma[o] + beta[o];
      dmat[lane * Oc + o] = (v >= 0.f) ? v : SLOPE * v;
    }
  }
  __syncthreads();
  for (int o = lane; o < Oc; o += 32) {    // max over k + residual
    float m = -3.4e38f;
#pragma unroll
    for (int r = 0; r < KSEL; ++r) m = fmaxf(m, dmat[r * Oc + o]);
    float rv = 0.f;
    if (RMODE == 1) rv = res[pbase * rstride + o];
    else if (RMODE == 2) {
#pragma unroll
      for (int c = 0; c < Cin; ++c)
        rv += align[o * Cin + c] * res[pbase * rstride + c];
    }
    out[pbase * ostride + o] = m + rv;
  }
}

// ---------------------------------------------------------------------------
// Kernel 4: KPConv layer (VALU, per-point workgroup of 64 threads)
// ---------------------------------------------------------------------------
template <int Cin, int Oc>
__global__ __launch_bounds__(64)
void kpconv_kernel(const float* __restrict__ pos, const int* __restrict__ idx,
                   int idx_stride,
                   const float* __restrict__ in, int istride,
                   const float* __restrict__ kpts, const float* __restrict__ kw,
                   const float* __restrict__ bias,
                   float* __restrict__ out, int ostride) {
  __shared__ int   idxl[KNBR];
  __shared__ float diff[KNBR * 3];
  __shared__ float infl[KNBR * NKP];
  __shared__ float nbrf[KNBR * Cin];
  __shared__ float agg[NKP * Cin];
  __shared__ float kpl[NKP * 3];
  const int tid = threadIdx.x;
  const int n = blockIdx.x % NPTS, b = blockIdx.x / NPTS;
  const size_t pbase = (size_t)b * NPTS + n;
  if (tid < NKP * 3) kpl[tid] = kpts[tid];
  if (tid < KNBR) idxl[tid] = idx[pbase * idx_stride + tid];
  __syncthreads();
  if (tid < KNBR) {
    int nb = idxl[tid];
#pragma unroll
    for (int c = 0; c < 3; ++c)
      diff[tid * 3 + c] = pos[((size_t)b * NPTS + nb) * 3 + c] - pos[pbase * 3 + c];
  }
  for (int e = tid; e < KNBR * Cin; e += 64) {
    int j = e / Cin, c = e % Cin;
    nbrf[j * Cin + c] = in[((size_t)b * NPTS + idxl[j]) * istride + c];
  }
  __syncthreads();
  for (int e = tid; e < KNBR * NKP; e += 64) {
    int j = e / NKP, i = e % NKP;
    float s = 0.f;
#pragma unroll
    for (int c = 0; c < 3; ++c) {
      float dd = diff[j * 3 + c] - kpl[i * 3 + c];
      s += dd * dd;
    }
    infl[e] = __expf(-s / 0.25f);          // SIGMA^2 = 0.25
  }
  __syncthreads();
  for (int e = tid; e < NKP * Cin; e += 64) {
    int i = e / Cin, c = e % Cin;
    float s = 0.f;
#pragma unroll
    for (int j = 0; j < KNBR; ++j) s += infl[j * NKP + i] * nbrf[j * Cin + c];
    agg[i * Cin + c] = s;
  }
  __syncthreads();
  for (int o = tid; o < Oc; o += 64) {
    float s = bias[o];
    for (int i = 0; i < NKP; ++i)
#pragma unroll
      for (int c = 0; c < Cin; ++c)
        s += agg[i * Cin + c] * kw[(i * Cin + c) * Oc + o];
    out[pbase * ostride + o] = fmaxf(s, 0.f);
  }
}

// ---------------------------------------------------------------------------
// Kernel 5: fusion GEMM (concat(feat_dg, feat_kp) @ W^T + b), WMMA 256x256
// ---------------------------------------------------------------------------
__global__ __launch_bounds__(32)
void fusion_kernel(const float* __restrict__ fdg, const float* __restrict__ fkp,
                   const float* __restrict__ W, const float* __restrict__ bias,
                   float* __restrict__ out) {
  __shared__ __align__(16) _Float16 inh[32 * 256];
  __shared__ __align__(16) _Float16 Wsl[16 * 256];
  const int lane = threadIdx.x;
  const int tiles = NPTS / 32;
  const int b = blockIdx.x / tiles;
  const int r0 = (blockIdx.x % tiles) * 32;
  {
    size_t base = (size_t)b * NPTS + r0 + lane;
    const float* pd = fdg + base * 128;
    const float* pk = fkp + base * 128;
#pragma unroll 16
    for (int c = 0; c < 128; ++c) {
      inh[lane * 256 + c]       = (_Float16)pd[c];
      inh[lane * 256 + 128 + c] = (_Float16)pk[c];
    }
  }
  for (int ot = 0; ot < 16; ++ot) {
    __syncthreads();
    for (int e = lane; e < 16 * 256; e += 32) {
      int o = e >> 8, c = e & 255;
      Wsl[e] = (_Float16)W[(ot * 16 + o) * 256 + c];
    }
    __syncthreads();
    v8f acc0 = {}, acc1 = {};
#pragma unroll
    for (int kc = 0; kc < 8; ++kc) {
      v16h a0 = frag_ld(inh, 0, 256, kc * 32, lane);
      v16h a1 = frag_ld(inh, 16, 256, kc * 32, lane);
      v16h bf = frag_ld(Wsl, 0, 256, kc * 32, lane);
      acc0 = WMMA(a0, bf, acc0);
      acc1 = WMMA(a1, bf, acc1);
    }
    const int col = lane & 15;
    const int rb  = (lane >> 4) << 3;
    const float bv = bias[ot * 16 + col];
#pragma unroll
    for (int r = 0; r < 8; ++r) {
      size_t row = (size_t)b * NPTS + r0 + rb + r;
      out[row * 256 + ot * 16 + col]        = acc0[r] + bv;
      out[(row + 16) * 256 + ot * 16 + col] = acc1[r] + bv;
    }
  }
}

// ---------------------------------------------------------------------------
// Kernel 6: PointTransformer layer (VALU, per-point workgroup of 64 threads)
// ---------------------------------------------------------------------------
template <int Cin, int D>
__global__ __launch_bounds__(64)
void pt_kernel(const float* __restrict__ pos, const int* __restrict__ idx,
               int idx_stride,
               const float* __restrict__ hin, int hstride,
               const float* __restrict__ Wc, const float* __restrict__ bc,
               const float* __restrict__ Wn, const float* __restrict__ bn,
               const float* __restrict__ pW1, const float* __restrict__ pb1,
               const float* __restrict__ pW2, const float* __restrict__ pb2,
               const float* __restrict__ aW1, const float* __restrict__ ab1,
               const float* __restrict__ aW2, const float* __restrict__ ab2,
               const float* __restrict__ sW, const float* __restrict__ sb,
               float* __restrict__ hout, int ostride) {
  __shared__ float hc[Cin];
  __shared__ float hn[KNBR * Cin];
  __shared__ float pd[KNBR * 3];
  __shared__ int   idxl[KNBR];
  __shared__ float fcv[D];
  __shared__ float fnv[KNBR * D];
  __shared__ float pev[KNBR * D];
  __shared__ float tA[KNBR * D];
  __shared__ float tB[KNBR * D];
  __shared__ float sc[KNBR];
  __shared__ float attn[KNBR];
  const int tid = threadIdx.x;
  const int n = blockIdx.x % NPTS, b = blockIdx.x / NPTS;
  const size_t pbase = (size_t)b * NPTS + n;
  for (int c = tid; c < Cin; c += 64) hc[c] = hin[pbase * hstride + c];
  if (tid < KNBR) idxl[tid] = idx[pbase * idx_stride + tid];
  __syncthreads();
  if (tid < KNBR) {
    int nb = idxl[tid];
#pragma unroll
    for (int c = 0; c < 3; ++c)
      pd[tid * 3 + c] = pos[pbase * 3 + c] - pos[((size_t)b * NPTS + nb) * 3 + c];
  }
  for (int e = tid; e < KNBR * Cin; e += 64) {
    int j = e / Cin, c = e % Cin;
    hn[j * Cin + c] = hin[((size_t)b * NPTS + idxl[j]) * hstride + c];
  }
  __syncthreads();
  for (int d = tid; d < D; d += 64) {          // fc = h @ Wc^T + bc
    float s = bc[d];
    for (int c = 0; c < Cin; ++c) s += Wc[d * Cin + c] * hc[c];
    fcv[d] = s;
  }
  for (int e = tid; e < KNBR * D; e += 64) {   // fn, pe1
    int j = e / D, d = e % D;
    float s = bn[d];
    for (int c = 0; c < Cin; ++c) s += Wn[d * Cin + c] * hn[j * Cin + c];
    fnv[j * D + d] = s;
    float p1 = pb1[d];
#pragma unroll
    for (int c = 0; c < 3; ++c) p1 += pW1[d * 3 + c] * pd[j * 3 + c];
    tA[j * D + d] = fmaxf(p1, 0.f);
  }
  __syncthreads();
  for (int e = tid; e < KNBR * D; e += 64) {   // pos_enc = relu(..)@pW2^T+pb2
    int j = e / D, d = e % D;
    float s = pb2[d];
#pragma unroll
    for (int c = 0; c < D; ++c) s += pW2[d * D + c] * tA[j * D + c];
    pev[j * D + d] = s;
  }
  __syncthreads();
  for (int e = tid; e < KNBR * D; e += 64) {   // t = fc - fn + pos_enc
    int j = e / D, d = e % D;
    tB[j * D + d] = fcv[d] - fnv[j * D + d] + pev[j * D + d];
  }
  __syncthreads();
  for (int e = tid; e < KNBR * D; e += 64) {   // a1 = relu(t@aW1^T+ab1)
    int j = e / D, d = e % D;
    float s = ab1[d];
#pragma unroll
    for (int c = 0; c < D; ++c) s += aW1[d * D + c] * tB[j * D + c];
    tA[j * D + d] = fmaxf(s, 0.f);
  }
  __syncthreads();
  for (int e = tid; e < KNBR * D; e += 64) {   // a2 = a1@aW2^T+ab2
    int j = e / D, d = e % D;
    float s = ab2[d];
#pragma unroll
    for (int c = 0; c < D; ++c) s += aW2[d * D + c] * tA[j * D + c];
    tB[j * D + d] = s;
  }
  __syncthreads();
  if (tid < KNBR) {                            // score
    float s = sb[0];
#pragma unroll
    for (int d = 0; d < D; ++d) s += sW[d] * tB[tid * D + d];
    sc[tid] = s;
  }
  __syncthreads();
  if (tid == 0) {                              // softmax over k
    float mx = sc[0];
#pragma unroll
    for (int j = 1; j < KNBR; ++j) mx = fmaxf(mx, sc[j]);
    float sum = 0.f;
#pragma unroll
    for (int j = 0; j < KNBR; ++j) { attn[j] = __expf(sc[j] - mx); sum += attn[j]; }
#pragma unroll
    for (int j = 0; j < KNBR; ++j) attn[j] /= sum;
  }
  __syncthreads();
  for (int d = tid; d < D; d += 64) {
    float s = fcv[d];
#pragma unroll
    for (int j = 0; j < KNBR; ++j) s += attn[j] * (fnv[j * D + d] + pev[j * D + d]);
    hout[pbase * ostride + d] = s;
  }
}

// ---------------------------------------------------------------------------
// Kernel 7: max+mean pooling over N for one channel per block
// ---------------------------------------------------------------------------
__global__ __launch_bounds__(128)
void pool_kernel(const float* __restrict__ f, int stride, int C,
                 float* __restrict__ dst, int off) {
  __shared__ float rmax[128], rsum[128];
  const int b = blockIdx.x / C, c = blockIdx.x % C;
  const int tid = threadIdx.x;
  float mx = -3.4e38f, sm = 0.f;
  for (int n = tid; n < NPTS; n += 128) {
    float v = f[((size_t)b * NPTS + n) * stride + c];
    mx = fmaxf(mx, v); sm += v;
  }
  rmax[tid] = mx; rsum[tid] = sm; __syncthreads();
  for (int o = 64; o > 0; o >>= 1) {
    if (tid < o) {
      rmax[tid] = fmaxf(rmax[tid], rmax[tid + o]);
      rsum[tid] += rsum[tid + o];
    }
    __syncthreads();
  }
  if (tid == 0) {
    dst[b * 640 + off + c]     = rmax[0];
    dst[b * 640 + off + C + c] = rsum[0] / (float)NPTS;
  }
}

// ---------------------------------------------------------------------------
// Kernel 8: final fc  out[b,o] = fc_b[o] + fc_W[o,:] . pooled[b,:]
// ---------------------------------------------------------------------------
__global__ __launch_bounds__(256)
void fc_kernel(const float* __restrict__ pooled, const float* __restrict__ W,
               const float* __restrict__ bias, float* __restrict__ out) {
  const int b = blockIdx.x, o = threadIdx.x;
  float s = bias[o];
  for (int i = 0; i < 640; ++i) s += W[o * 640 + i] * pooled[b * 640 + i];
  out[b * 256 + o] = s;
}

// ---------------------------------------------------------------------------
// Launch. Input leaf order assumes JAX pytree flatten (sorted dict keys,
// uppercase < lowercase):
//  params.dgcnn: W1 W2 W3 W4 align1 align2 b1..b4 g1..g4        -> 0..13
//  params.fc_W 14, fc_b 15, fusion_W 16, fusion_b 17
//  params.kpconv[l]: bias, kernel_points, kw                    -> 18+3l..
//  params.pt[l]: Wc Wn aW1 aW2 ab1 ab2 bc bn pW1 pW2 pb1 pb2 sW sb -> 30+14l..
//  x                                                            -> 86
// ---------------------------------------------------------------------------
extern "C" void kernel_launch(void* const* d_in, const int* in_sizes, int n_in,
                              void* d_out, int out_size, void* d_ws, size_t ws_size,
                              hipStream_t stream) {
  (void)in_sizes; (void)n_in; (void)out_size; (void)ws_size;
  auto F = [&](int i) { return (const float*)d_in[i]; };
  const int B = BATCH, N = NPTS;

  const float* dgW[4]  = {F(0), F(1), F(2), F(3)};
  const float* align1 = F(4); const float* align2 = F(5);
  const float* dgb[4] = {F(6), F(7), F(8), F(9)};
  const float* dgg[4] = {F(10), F(11), F(12), F(13)};
  const float* fcW = F(14); const float* fcb = F(15);
  const float* fuW = F(16); const float* fub = F(17);
  const float *kpBias[4], *kpPts[4], *kpKw[4];
  for (int l = 0; l < 4; ++l) {
    kpBias[l] = F(18 + 3 * l); kpPts[l] = F(19 + 3 * l); kpKw[l] = F(20 + 3 * l);
  }
  const float* pt[4][14];
  for (int l = 0; l < 4; ++l)
    for (int j = 0; j < 14; ++j) pt[l][j] = F(30 + 14 * l + j);
  // pt leaf index: 0 Wc, 1 Wn, 2 aW1, 3 aW2, 4 ab1, 5 ab2, 6 bc, 7 bn,
  //                8 pW1, 9 pW2, 10 pb1, 11 pb2, 12 sW, 13 sb
  const float* x = F(86);

  // workspace layout
  float* pos    = (float*)d_ws;                             // B*N*3
  int*   idxP   = (int*)(pos + (size_t)B * N * 3);          // B*N*20 (pos knn)
  int*   idxT   = idxP + (size_t)B * N * KSEL;              // B*N*20 (reused)
  float* fdg    = (float*)(idxT + (size_t)B * N * KSEL);    // B*N*128
  float* fkp    = fdg + (size_t)B * N * 128;                // B*N*128
  float* fused  = fkp + (size_t)B * N * 128;                // B*N*256
  float* hA     = fused + (size_t)B * N * 256;              // B*N*64
  float* hB     = hA + (size_t)B * N * 64;                  // B*N*64
  float* pooled = hB + (size_t)B * N * 64;                  // B*640

  const int knnGrid = B * (N / 32);
  const int ptGrid  = B * N;

  normalize_kernel<<<B, 256, 0, stream>>>(x, pos);

  // ---- DGCNN branch (outputs into slices of fdg[.,.,128]) ----
  knn_kernel<3><<<knnGrid, 32, 0, stream>>>(pos, 3, idxP, KSEL);
  edgeconv_kernel<3, 16, 0><<<ptGrid, 32, 0, stream>>>(pos, 3, idxP, KSEL,
      dgW[0], dgg[0], dgb[0], nullptr, 0, nullptr, fdg + 0, 128);
  knn_kernel<16><<<knnGrid, 32, 0, stream>>>(fdg + 0, 128, idxT, KSEL);
  edgeconv_kernel<16, 16, 1><<<ptGrid, 32, 0, stream>>>(fdg + 0, 128, idxT, KSEL,
      dgW[1], dgg[1], dgb[1], fdg + 0, 128, nullptr, fdg + 16, 128);
  knn_kernel<16><<<knnGrid, 32, 0, stream>>>(fdg + 16, 128, idxT, KSEL);
  edgeconv_kernel<16, 32, 2><<<ptGrid, 32, 0, stream>>>(fdg + 16, 128, idxT, KSEL,
      dgW[2], dgg[2], dgb[2], fdg + 16, 128, align1, fdg + 32, 128);
  knn_kernel<32><<<knnGrid, 32, 0, stream>>>(fdg + 32, 128, idxT, KSEL);
  edgeconv_kernel<32, 64, 2><<<ptGrid, 32, 0, stream>>>(fdg + 32, 128, idxT, KSEL,
      dgW[3], dgg[3], dgb[3], fdg + 32, 128, align2, fdg + 64, 128);

  // ---- KPConv branch (uses first 16 of pos-knn; slices of fkp) ----
  kpconv_kernel<128, 16><<<ptGrid, 64, 0, stream>>>(pos, idxP, KSEL, fdg, 128,
      kpPts[0], kpKw[0], kpBias[0], fkp + 0, 128);
  kpconv_kernel<16, 16><<<ptGrid, 64, 0, stream>>>(pos, idxP, KSEL, fkp + 0, 128,
      kpPts[1], kpKw[1], kpBias[1], fkp + 16, 128);
  kpconv_kernel<16, 32><<<ptGrid, 64, 0, stream>>>(pos, idxP, KSEL, fkp + 16, 128,
      kpPts[2], kpKw[2], kpBias[2], fkp + 32, 128);
  kpconv_kernel<32, 64><<<ptGrid, 64, 0, stream>>>(pos, idxP, KSEL, fkp + 32, 128,
      kpPts[3], kpKw[3], kpBias[3], fkp + 64, 128);

  // ---- fusion GEMM ----
  fusion_kernel<<<B * (N / 32), 32, 0, stream>>>(fdg, fkp, fuW, fub, fused);

  // ---- point transformer stack ----
  {
    const float* p0[14]; const float* p1[14]; const float* p2[14]; const float* p3[14];
    for (int j = 0; j < 14; ++j) { p0[j]=pt[0][j]; p1[j]=pt[1][j]; p2[j]=pt[2][j]; p3[j]=pt[3][j]; }
    pt_kernel<256, 16><<<ptGrid, 64, 0, stream>>>(pos, idxP, KSEL, fused, 256,
        p0[0], p0[6], p0[1], p0[7], p0[8], p0[10], p0[9], p0[11],
        p0[2], p0[4], p0[3], p0[5], p0[12], p0[13], hA, 16);
    pt_kernel<16, 16><<<ptGrid, 64, 0, stream>>>(pos, idxP, KSEL, hA, 16,
        p1[0], p1[6], p1[1], p1[7], p1[8], p1[10], p1[9], p1[11],
        p1[2], p1[4], p1[3], p1[5], p1[12], p1[13], hB, 16);
    pt_kernel<16, 32><<<ptGrid, 64, 0, stream>>>(pos, idxP, KSEL, hB, 16,
        p2[0], p2[6], p2[1], p2[7], p2[8], p2[10], p2[9], p2[11],
        p2[2], p2[4], p2[3], p2[5], p2[12], p2[13], hA, 32);
    pt_kernel<32, 64><<<ptGrid, 64, 0, stream>>>(pos, idxP, KSEL, hA, 32,
        p3[0], p3[6], p3[1], p3[7], p3[8], p3[10], p3[9], p3[11],
        p3[2], p3[4], p3[3], p3[5], p3[12], p3[13], hB, 64);
  }

  // ---- pooling heads: [pool(h)(128) | pool(fdg)(256) | pool(fkp)(256)] ----
  pool_kernel<<<B * 64, 128, 0, stream>>>(hB, 64, 64, pooled, 0);
  pool_kernel<<<B * 128, 128, 0, stream>>>(fdg, 128, 128, pooled, 128);
  pool_kernel<<<B * 128, 128, 0, stream>>>(fkp, 128, 128, pooled, 384);

  fc_kernel<<<B, 256, 0, stream>>>(pooled, fcW, fcb, (float*)d_out);
}
